// WaterGraphNet_85899345920547
// MI455X (gfx1250) — matrix operand order repaired and checked by reading, not compiled, not run on hardware
//
#include <hip/hip_runtime.h>

// ---------------------------------------------------------------------------
// WaterGraphNet for MI455X (gfx1250): series-decompose + 2x SAGE res-blocks.
// bf16 WMMA (v_wmma_f32_16x16x32_bf16) for all GEMMs, f32 L2 atomics for the
// graph mean-aggregation, BN stats fused into the GEMM epilogue.
// GEMM A-tiles are double-buffered in LDS; the tile copy uses the gfx1250
// GLOBAL_LOAD_ASYNC_TO_LDS_B128 path (ASYNCcnt) when the builtin is exposed,
// falling back to a VGPR-staged copy otherwise.
// ---------------------------------------------------------------------------

typedef __attribute__((ext_vector_type(16))) __bf16 v16bf;
typedef __attribute__((ext_vector_type(8)))  __bf16 v8bf;
typedef __attribute__((ext_vector_type(8)))  float  v8f;
typedef __attribute__((ext_vector_type(4)))  unsigned int v4u;
typedef __attribute__((ext_vector_type(4)))  int v4i;

#define IN_C  128
#define OUT_C 64
#define KW    25
#define PADW  12

#if defined(__gfx1250__) && __has_builtin(__builtin_amdgcn_global_load_async_to_lds_b128) && \
    __has_builtin(__builtin_amdgcn_s_wait_asynccnt)
#define HAVE_ASYNC_LDS 1
#else
#define HAVE_ASYNC_LDS 0
#endif

#if HAVE_ASYNC_LDS
typedef __attribute__((address_space(1))) v4i gv4i;   // global v4i
typedef __attribute__((address_space(3))) v4i lv4i;   // LDS v4i
__device__ __forceinline__ void async_cp16(__bf16* lptr, const __bf16* gptr) {
  __builtin_amdgcn_global_load_async_to_lds_b128(
      (gv4i*)gptr, (lv4i*)lptr, /*offset=*/0, /*cpol=*/0);
}
#endif

// ---------------- zero workspace (atomic accumulators + stats) -------------
__global__ void k_zero(float* __restrict__ p, int n) {
  int i = blockIdx.x * blockDim.x + threadIdx.x;
  int stride = gridDim.x * blockDim.x;
  for (; i < n; i += stride) p[i] = 0.f;
}

// ---------------- decompose: trend = sliding mean(25, replicate pad) -------
__global__ void k_decompose(const float* __restrict__ x,
                            __bf16* __restrict__ cat_s,
                            __bf16* __restrict__ cat_t, int N) {
  __shared__ float row[IN_C];
  const int n = blockIdx.x;
  const int c = threadIdx.x;
  row[c] = x[(size_t)n * IN_C + c];
  __syncthreads();
  float s = 0.f;
#pragma unroll
  for (int d = -PADW; d <= PADW; ++d) {
    int j = c + d;
    j = j < 0 ? 0 : (j > IN_C - 1 ? IN_C - 1 : j);
    s += row[j];
  }
  const float trend  = s * (1.0f / (float)KW);
  const float season = row[c] - trend;
  cat_t[(size_t)n * (2 * IN_C) + IN_C + c] = (__bf16)trend;
  cat_s[(size_t)n * (2 * IN_C) + IN_C + c] = (__bf16)season;
}

// ---------------- pack weights into WMMA B-fragment layout -----------------
// B[k][m]; k<K/2 -> Wl (aggregated input), k>=K/2 -> Wr (self input).
// m<64 -> conv A (c1), m>=64 -> conv B (c3).  Fragment: 32 lanes x 16 bf16
// contiguous; lane<16 -> col nt*16+lane, k = kt*32+e; lane>=16 -> k += 16.
__global__ void k_pack(const float* __restrict__ WlA, const float* __restrict__ WrA,
                       const float* __restrict__ WlB, const float* __restrict__ WrB,
                       __bf16* __restrict__ out, int K, int M) {
  int idx = blockIdx.x * blockDim.x + threadIdx.x;
  if (idx >= K * M) return;
  int f = idx >> 9;            // fragment index (512 elems each)
  int r = idx & 511;
  int lane = r >> 4, e = r & 15;
  int NTl = M >> 4;
  int kt = f / NTl, nt = f % NTl;
  int col = nt * 16 + (lane & 15);
  int kk  = kt * 32 + ((lane >> 4) << 4) + e;
  int kh  = K >> 1;
  const float *Wl, *Wr;
  int m = col;
  if (m < 64) { Wl = WlA; Wr = WrA; } else { Wl = WlB; Wr = WrB; m -= 64; }
  float w = (kk < kh) ? Wl[(size_t)m * kh + kk] : Wr[(size_t)m * kh + (kk - kh)];
  out[idx] = (__bf16)w;
}

// ---------------- edge aggregation: wave per edge, f32 atomics in L2 -------
__global__ void k_edge_agg(const int* __restrict__ srcIdx, const int* __restrict__ dstIdx,
                           int E,
                           const __bf16* __restrict__ X0, const __bf16* __restrict__ X1,
                           int ld, int coff, int C,
                           float* __restrict__ A0, float* __restrict__ A1,
                           float* __restrict__ deg) {
  const int wave = (blockIdx.x * blockDim.x + threadIdx.x) >> 5;
  const int lane = threadIdx.x & 31;
  if (wave >= E) return;
  const int s = srcIdx[wave];
  const int d = dstIdx[wave];
  const __bf16* x0 = X0 + (size_t)s * ld + coff;
  const __bf16* x1 = X1 + (size_t)s * ld + coff;
  float* a0 = A0 + (size_t)d * C;
  float* a1 = A1 + (size_t)d * C;
  for (int c = lane; c < C; c += 32) {
    atomicAdd(&a0[c], (float)x0[c]);
    atomicAdd(&a1[c], (float)x1[c]);
  }
  if (deg != nullptr && lane == 0) atomicAdd(&deg[d], 1.0f);
}

// ---------------- divide by degree, convert to bf16 into cat buffer --------
__global__ void k_div_bf(const float* __restrict__ Agg, const float* __restrict__ deg,
                         __bf16* __restrict__ out, int N, int C, int ldo, int ooff) {
  int i = blockIdx.x * blockDim.x + threadIdx.x;
  if (i >= N * C) return;
  int n = i / C, c = i % C;
  float dd = fmaxf(deg[n], 1.0f);
  out[(size_t)n * ldo + ooff + c] = (__bf16)(Agg[i] / dd);
}

// ---------------- WMMA GEMM: [N,K]bf16 x [K,M]bf16 -> [N,M]f32 (+BN stats) -
// 256 threads = 8 waves; 64 rows per block; A double-buffered in LDS
// (padded stride), B read from the packed fragment buffer (32B/lane).
// One workgroup barrier per k-step; next tile copy overlaps the WMMA chain.
template <int K, int M>
__global__ void __launch_bounds__(256)
gemm_bn(const __bf16* __restrict__ A, const __bf16* __restrict__ Bpk,
        float* __restrict__ Out, float* __restrict__ statSum,
        float* __restrict__ statSq, int N) {
  constexpr int NT  = M / 16;   // output col tiles
  constexpr int KT  = K / 32;   // k steps
  constexpr int TPW = NT / 2;   // col tiles per wave
  constexpr int LROW = 40;      // padded LDS row stride (bf16 elems)
  __shared__ __bf16 lds[2][64 * LROW];

  const int tid  = threadIdx.x;
  const int w    = tid >> 5;
  const int lane = tid & 31;
  const int rowBase = blockIdx.x * 64;
  const int rt = w >> 1;              // row tile 0..3
  const int nb = (w & 1) * TPW;       // first col tile

  v8f acc[TPW] = {};

  // cooperative A-tile copy mapping (64 rows x 32 k, 16B per thread)
  const int lrow   = tid >> 2;
  const int lchunk = tid & 3;
  int grow = rowBase + lrow;
  if (grow > N - 1) grow = N - 1;     // clamp; bogus rows masked at store
  const __bf16* gsrc = A + (size_t)grow * K + lchunk * 8;
  const int ldsOff = lrow * LROW + lchunk * 8;

  // A-fragment read mapping (lane<16: k 0..15; lane>=16: k 16..31)
  const int arow = rt * 16 + (lane & 15);
  const int kl0  = (lane >> 4) * 16;
  const int afrOff = arow * LROW + kl0;

#if HAVE_ASYNC_LDS
  // prologue: async copy of tile 0 straight into LDS (ASYNCcnt tracked)
  async_cp16(&lds[0][ldsOff], gsrc);
#else
  {
    v4u stage = *(const v4u*)gsrc;
    *(v4u*)&lds[0][ldsOff] = stage;
  }
#endif

  for (int kt = 0; kt < KT; ++kt) {
    const int cur = kt & 1, nxt = cur ^ 1;
#if HAVE_ASYNC_LDS
    __builtin_amdgcn_s_wait_asynccnt(0);   // own tile-copy landed in LDS
    __syncthreads();                       // everyone's copy landed; prior reads done
    if (kt + 1 < KT)                       // overlap next copy with WMMA chain
      async_cp16(&lds[nxt][ldsOff], gsrc + (kt + 1) * 32);
#else
    __syncthreads();
    v4u stage;
    if (kt + 1 < KT) stage = *(const v4u*)(gsrc + (kt + 1) * 32);  // issue early
#endif

    union { v16bf v; v8bf h[2]; } au;
    au.h[0] = *(const v8bf*)&lds[cur][afrOff];
    au.h[1] = *(const v8bf*)&lds[cur][afrOff + 8];
#pragma unroll
    for (int t = 0; t < TPW; ++t) {
      const v16bf bf = *(const v16bf*)(Bpk + ((size_t)(kt * NT + nb + t) * 32 + lane) * 16);
      acc[t] = __builtin_amdgcn_wmma_f32_16x16x32_bf16(
          false, au.v, false, bf, (short)0, acc[t], false, false);
    }

#if !HAVE_ASYNC_LDS
    if (kt + 1 < KT) *(v4u*)&lds[nxt][ldsOff] = stage;  // after compute; synced at loop top
#endif
  }

  // store + fused BN statistics (per-column sum / sumsq over valid rows)
  const int rstore = rowBase + rt * 16 + ((lane >> 4) << 3);
#pragma unroll
  for (int t = 0; t < TPW; ++t) {
    const int col = (nb + t) * 16 + (lane & 15);
    float s = 0.f, q = 0.f;
#pragma unroll
    for (int r = 0; r < 8; ++r) {
      const int row = rstore + r;
      const float v = acc[t][r];
      if (row < N) {
        Out[(size_t)row * M + col] = v;
        s += v; q += v * v;
      }
    }
    s += __shfl_xor(s, 16);
    q += __shfl_xor(q, 16);
    if (lane < 16) {
      atomicAdd(&statSum[col], s);
      atomicAdd(&statSq[col], q);
    }
  }
}

// ---------------- BN stats -> affine coeffs (a = g*rsqrt(var+eps), c) ------
__global__ void k_bn_coef(const float* __restrict__ g, const float* __restrict__ b,
                          const float* __restrict__ sum, const float* __restrict__ sq,
                          float* __restrict__ coef, float invN) {
  int o = threadIdx.x;
  if (o >= OUT_C) return;
  float m   = sum[o] * invN;
  float var = sq[o] * invN - m * m;
  float a   = g[o] * rsqrtf(var + 1e-5f);
  coef[o]         = a;
  coef[OUT_C + o] = b[o] - m * a;     // fold mean (and cancelled conv bias)
}

// ---------------- BN + ReLU -> bf16 (self half of GEMM2 input) -------------
__global__ void k_bn_relu(const float* __restrict__ pre, const float* __restrict__ coefL,
                          __bf16* __restrict__ out, int N) {
  int i = blockIdx.x * blockDim.x + threadIdx.x;
  if (i >= N * OUT_C) return;
  int n = i >> 6, o = i & 63;
  float v = fmaf(pre[(size_t)n * 128 + o], coefL[o], coefL[OUT_C + o]);
  out[(size_t)n * 128 + OUT_C + o] = (__bf16)fmaxf(v, 0.f);
}

// ---------------- final: relu(bn(h2)+bn(xr)) per block, summed -------------
__global__ void k_final(const float* __restrict__ h2s, const float* __restrict__ h2t,
                        const float* __restrict__ pre1s, const float* __restrict__ pre1t,
                        const float* __restrict__ coef, float* __restrict__ out, int N) {
  int i = blockIdx.x * blockDim.x + threadIdx.x;
  if (i >= N * OUT_C) return;
  int n = i >> 6, o = i & 63;
  // coef layers (stride 128): 0=n1s 1=n3s 2=n1t 3=n3t 4=n2s 5=n2t
  float vs = coef[4 * 128 + o] * h2s[i] + coef[4 * 128 + 64 + o]
           + coef[1 * 128 + o] * pre1s[(size_t)n * 128 + 64 + o] + coef[1 * 128 + 64 + o];
  float vt = coef[5 * 128 + o] * h2t[i] + coef[5 * 128 + 64 + o]
           + coef[3 * 128 + o] * pre1t[(size_t)n * 128 + 64 + o] + coef[3 * 128 + 64 + o];
  out[i] = fmaxf(vs, 0.f) + fmaxf(vt, 0.f);
}

// ---------------------------------------------------------------------------
extern "C" void kernel_launch(void* const* d_in, const int* in_sizes, int n_in,
                              void* d_out, int out_size, void* d_ws, size_t ws_size,
                              hipStream_t stream) {
  (void)n_in; (void)out_size; (void)ws_size;
  const float* x = (const float*)d_in[0];
  const int*   ei = (const int*)d_in[1];
  const int N = in_sizes[0] / IN_C;
  const int E = in_sizes[1] / 2;
  const int* srcI = ei;
  const int* dstI = ei + E;
#define P(i) ((const float*)d_in[(i)])
  // params (insertion order): sage1 @2: c1.Wl,Wr,bl, n1.g,b, c2.Wl,Wr,bl,
  // n2.g,b, c3.Wl,Wr,bl, n3.g,b ; sage2 @17 same. (conv biases cancel in BN)

  // ---- workspace carve-out (256B aligned) ----
  char* p = (char*)d_ws;
  auto alloc = [&](size_t bytes) { char* r = p; p += (bytes + 255) & ~(size_t)255; return r; };
  float* AggS  = (float*)alloc((size_t)N * 128 * 4);   // zeroed region start
  float* AggT  = (float*)alloc((size_t)N * 128 * 4);
  float* AggHs = (float*)alloc((size_t)N * 64 * 4);
  float* AggHt = (float*)alloc((size_t)N * 64 * 4);
  float* deg   = (float*)alloc((size_t)N * 4);
  float* stats = (float*)alloc(768 * 4);               // 6 layers x (sum|sq)
  const size_t zbytes = (size_t)(p - (char*)d_ws);     // zeroed region end
  __bf16* cat_s  = (__bf16*)alloc((size_t)N * 256 * 2); // [aggS | s]
  __bf16* cat_t  = (__bf16*)alloc((size_t)N * 256 * 2); // [aggT | t]
  __bf16* cat2_s = (__bf16*)alloc((size_t)N * 128 * 2); // [aggH1s | h1s]
  __bf16* cat2_t = (__bf16*)alloc((size_t)N * 128 * 2);
  float* pre1s = (float*)alloc((size_t)N * 128 * 4);    // cols 0-63 h1pre, 64-127 xrpre
  float* pre1t = (float*)alloc((size_t)N * 128 * 4);
  float* h2s   = (float*)alloc((size_t)N * 64 * 4);
  float* h2t   = (float*)alloc((size_t)N * 64 * 4);
  float* coef  = (float*)alloc(6 * 128 * 4);
  __bf16* W1s = (__bf16*)alloc(256 * 128 * 2);
  __bf16* W1t = (__bf16*)alloc(256 * 128 * 2);
  __bf16* W2s = (__bf16*)alloc(128 * 64 * 2);
  __bf16* W2t = (__bf16*)alloc(128 * 64 * 2);

  const int nz = (int)(zbytes / 4);
  k_zero<<<dim3((nz + 255) / 256), dim3(256), 0, stream>>>((float*)d_ws, nz);

  k_decompose<<<dim3(N), dim3(IN_C), 0, stream>>>(x, cat_s, cat_t, N);

  k_pack<<<dim3((256 * 128 + 255) / 256), 256, 0, stream>>>(P(2), P(3), P(12), P(13), W1s, 256, 128);
  k_pack<<<dim3((256 * 128 + 255) / 256), 256, 0, stream>>>(P(17), P(18), P(27), P(28), W1t, 256, 128);
  k_pack<<<dim3((128 * 64 + 255) / 256), 256, 0, stream>>>(P(7), P(8), P(7), P(8), W2s, 128, 64);
  k_pack<<<dim3((128 * 64 + 255) / 256), 256, 0, stream>>>(P(22), P(23), P(22), P(23), W2t, 128, 64);

  // aggregate seasonal & trend (128 ch each) + degrees
  k_edge_agg<<<dim3((E + 7) / 8), 256, 0, stream>>>(srcI, dstI, E, cat_s, cat_t,
                                                    256, 128, 128, AggS, AggT, deg);
  k_div_bf<<<dim3((N * 128 + 255) / 256), 256, 0, stream>>>(AggS, deg, cat_s, N, 128, 256, 0);
  k_div_bf<<<dim3((N * 128 + 255) / 256), 256, 0, stream>>>(AggT, deg, cat_t, N, 128, 256, 0);

  // GEMM1: [agg|self](N,256) x (256,128) -> [h1pre|xrpre]; BN stats fused
  const dim3 gg((N + 63) / 64);
  gemm_bn<256, 128><<<gg, 256, 0, stream>>>(cat_s, W1s, pre1s, stats + 0,   stats + 128, N);
  gemm_bn<256, 128><<<gg, 256, 0, stream>>>(cat_t, W1t, pre1t, stats + 256, stats + 384, N);

  const float invN = 1.0f / (float)N;
  k_bn_coef<<<1, 64, 0, stream>>>(P(5),  P(6),  stats + 0,   stats + 128, coef + 0,   invN); // n1s
  k_bn_coef<<<1, 64, 0, stream>>>(P(15), P(16), stats + 64,  stats + 192, coef + 128, invN); // n3s
  k_bn_coef<<<1, 64, 0, stream>>>(P(20), P(21), stats + 256, stats + 384, coef + 256, invN); // n1t
  k_bn_coef<<<1, 64, 0, stream>>>(P(30), P(31), stats + 320, stats + 448, coef + 384, invN); // n3t

  k_bn_relu<<<dim3((N * 64 + 255) / 256), 256, 0, stream>>>(pre1s, coef + 0,   cat2_s, N);
  k_bn_relu<<<dim3((N * 64 + 255) / 256), 256, 0, stream>>>(pre1t, coef + 256, cat2_t, N);

  // aggregate h1 (64 ch each block)
  k_edge_agg<<<dim3((E + 7) / 8), 256, 0, stream>>>(srcI, dstI, E, cat2_s, cat2_t,
                                                    128, 64, 64, AggHs, AggHt, nullptr);
  k_div_bf<<<dim3((N * 64 + 255) / 256), 256, 0, stream>>>(AggHs, deg, cat2_s, N, 64, 128, 0);
  k_div_bf<<<dim3((N * 64 + 255) / 256), 256, 0, stream>>>(AggHt, deg, cat2_t, N, 64, 128, 0);

  // GEMM2: [aggH|h1](N,128) x (128,64) -> h2pre; BN stats fused
  gemm_bn<128, 64><<<gg, 256, 0, stream>>>(cat2_s, W2s, h2s, stats + 512, stats + 576, N);
  gemm_bn<128, 64><<<gg, 256, 0, stream>>>(cat2_t, W2t, h2t, stats + 640, stats + 704, N);

  k_bn_coef<<<1, 64, 0, stream>>>(P(10), P(11), stats + 512, stats + 576, coef + 512, invN); // n2s
  k_bn_coef<<<1, 64, 0, stream>>>(P(25), P(26), stats + 640, stats + 704, coef + 640, invN); // n2t

  k_final<<<dim3((N * 64 + 255) / 256), 256, 0, stream>>>(h2s, h2t, pre1s, pre1t, coef,
                                                          (float*)d_out, N);
#undef P
}
